// TensorParallelAttention_80796924772820
// MI455X (gfx1250) — matrix operand order repaired and checked by reading, not compiled, not run on hardware
//
#include <hip/hip_runtime.h>
#include <hip/hip_bf16.h>
#include <stdint.h>

// ---------------------------------------------------------------------------
// TensorParallelAttention for MI455X (gfx1250, wave32, WMMA).
// Pipeline: f32->bf16 convert, QKV GEMM (WMMA bf16), flash attention
// (WMMA bf16 QK^T and PV, fp32 online softmax), output GEMM (WMMA bf16).
// Round 4: LDS double buffering in both GEMM and attention so the async
// global->LDS copies (ASYNCcnt) and the VGPR-staged transpose loads overlap
// with the WMMA compute of the current tile; single barrier per iteration.
// ---------------------------------------------------------------------------

typedef __bf16 bf16_t;
typedef __attribute__((ext_vector_type(16))) __bf16 bf16x16;
typedef __attribute__((ext_vector_type(8)))  __bf16 bf16x8;
typedef __attribute__((ext_vector_type(8)))  float  f32x8;
typedef __attribute__((ext_vector_type(4)))  int    i32x4;

#define HID   1024
#define NH    16
#define NKV   4
#define HD    64
#define BATCH 2
#define SEQ   2048
#define MTOT  (BATCH * SEQ)            // 4096 rows
#define NQKV  ((NH + 2 * NKV) * HD)    // 1536
#define ATT_SCALE 0.125f               // 1/sqrt(64)

#define WMMA_BF16(a, b, c) \
  __builtin_amdgcn_wmma_f32_16x16x32_bf16(false, (a), false, (b), (short)0, (c), false, false)

#if defined(__gfx1250__) && __has_builtin(__builtin_amdgcn_global_load_async_to_lds_b128) && \
    __has_builtin(__builtin_amdgcn_s_wait_asynccnt)
#define HAVE_ASYNC_LDS 1
#else
#define HAVE_ASYNC_LDS 0
#endif

#define AS_GLOBAL __attribute__((address_space(1)))
#define AS_LDS    __attribute__((address_space(3)))

// Copy 16 bytes global -> LDS. Async (ASYNCcnt) when available, else via VGPRs.
static __device__ __forceinline__ void copy16_g2l(const bf16_t* gsrc, bf16_t* ldst) {
#if HAVE_ASYNC_LDS
  __builtin_amdgcn_global_load_async_to_lds_b128(
      (AS_GLOBAL i32x4*)(bf16_t*)gsrc, (AS_LDS i32x4*)ldst, 0, 0);
#else
  *(uint4*)ldst = *(const uint4*)gsrc;
#endif
}
static __device__ __forceinline__ void copy_wait() {
#if HAVE_ASYNC_LDS
  __builtin_amdgcn_s_wait_asynccnt(0);
#endif
}

// A-fragment (16x32 bf16, M = lane%16):
//   lanes 0-15 : elems 0-7 -> K = 0..7,   elems 8-15 -> K = 16..23
//   lanes16-31 : elems 0-7 -> K = 8..15,  elems 8-15 -> K = 24..31
// `base` points at (row, K=0) of the current 32-wide K slice (K-contiguous).
static __device__ __forceinline__ bf16x16 ld_frag_a(const bf16_t* base, int hi) {
  bf16x8 lo  = *(const bf16x8*)(base + hi * 8);
  bf16x8 hiv = *(const bf16x8*)(base + 16 + hi * 8);
  return __builtin_shufflevector(lo, hiv, 0, 1, 2, 3, 4, 5, 6, 7, 8, 9, 10, 11, 12, 13, 14, 15);
}

// B-fragment (32x16 bf16, N = lane%16):
//   lanes 0-15 : elems 0-7 -> K = 0..7, elems 8-15 -> K = 16..23 ; lanes 16-31 : K = 8..15, 24..31
// `colbase` points at (col, K=0) of the current 32-wide K slice (K-contiguous).
static __device__ __forceinline__ bf16x16 ld_frag_b(const bf16_t* colbase, int hi) {
  bf16x8 lo  = *(const bf16x8*)(colbase + hi * 8);
  bf16x8 hiv = *(const bf16x8*)(colbase + 16 + hi * 8);
  return __builtin_shufflevector(lo, hiv, 0, 1, 2, 3, 4, 5, 6, 7, 8, 9, 10, 11, 12, 13, 14, 15);
}

// ---------------------------------------------------------------------------
// f32 -> bf16 elementwise convert
// ---------------------------------------------------------------------------
__global__ __launch_bounds__(256) void f32_to_bf16_kernel(const float* __restrict__ in,
                                                          bf16_t* __restrict__ out, int n) {
  int i = blockIdx.x * 256 + threadIdx.x;
  if (i < n) out[i] = (bf16_t)in[i];
}

// ---------------------------------------------------------------------------
// Tiled WMMA GEMM:  C[M x N] = A[M x HID] * B[HID x N]    (A,B bf16 row-major)
// Block tile 128(M) x 64(N), 8 waves, wave = 16 rows x 64 cols (4 accumulators).
// Double-buffered LDS: async A copy + B register load for slice i+1 are
// issued before the WMMA block of slice i.
// mode 0: scatter epilogue into Q/K/V bf16 [b, head, s, d] layouts (N = 1536)
// mode 1: plain fp32 store into Cout [M x 1024]            (N = 1024)
// ---------------------------------------------------------------------------
__global__ __launch_bounds__(256)
void gemm_bf16_wmma(const bf16_t* __restrict__ A, const bf16_t* __restrict__ Bm,
                    int N, int mode,
                    bf16_t* __restrict__ Qb, bf16_t* __restrict__ Kb, bf16_t* __restrict__ Vb,
                    float* __restrict__ Cout) {
  __shared__ __align__(16) bf16_t Ash[2][128 * 32];   // [row][k]  2 x 8 KB
  __shared__ __align__(16) bf16_t Bsh[2][64 * 32];    // [n][k]    2 x 4 KB

  const int tid  = threadIdx.x;
  const int wave = tid >> 5;
  const int lane = tid & 31;
  const int lr   = lane & 15;
  const int hi   = lane >> 4;
  const int m0   = blockIdx.x * 128;
  const int n0   = blockIdx.y * 64;

  const int arow = tid >> 1, akh = (tid & 1) * 16;   // A-tile staging role
  const int bk   = tid >> 3, bnb = (tid & 7) * 8;    // B-tile staging role

  // Prologue: stage k-slice 0 into buffer 0.
  {
    const bf16_t* s = A + (size_t)(m0 + arow) * HID + akh;
    copy16_g2l(s, &Ash[0][arow * 32 + akh]);
    copy16_g2l(s + 8, &Ash[0][arow * 32 + akh + 8]);
    bf16x8 v = *(const bf16x8*)(Bm + (size_t)bk * N + n0 + bnb);
#pragma unroll
    for (int j = 0; j < 8; ++j) Bsh[0][(bnb + j) * 32 + bk] = v[j];
  }
  copy_wait();
  __syncthreads();

  f32x8 acc[4] = {};

  for (int kk = 0; kk < HID; kk += 32) {
    const int cur = (kk >> 5) & 1, nxt = cur ^ 1;
    const bool has_next = (kk + 32 < HID);

    bf16x8 bv = {};
    if (has_next) {  // issue next-slice staging before computing on `cur`
      const bf16_t* s = A + (size_t)(m0 + arow) * HID + kk + 32 + akh;
      copy16_g2l(s, &Ash[nxt][arow * 32 + akh]);
      copy16_g2l(s + 8, &Ash[nxt][arow * 32 + akh + 8]);
      bv = *(const bf16x8*)(Bm + (size_t)(kk + 32 + bk) * N + n0 + bnb);
      if (kk + 64 < HID)  // two slices ahead -> global_prefetch_b8
        __builtin_prefetch(Bm + (size_t)(kk + 64) * N + n0 + (tid & 63), 0, 3);
    }

    // WMMA block on the current buffer (overlaps async copy + global loads)
    bf16x16 a = ld_frag_a(&Ash[cur][(wave * 16 + lr) * 32], hi);
#pragma unroll
    for (int t = 0; t < 4; ++t) {
      bf16x16 b = ld_frag_b(&Bsh[cur][(t * 16 + lr) * 32], hi);
      acc[t] = WMMA_BF16(a, b, acc[t]);
    }

    if (has_next) {  // transpose-store B slice into the other buffer
#pragma unroll
      for (int j = 0; j < 8; ++j) Bsh[nxt][(bnb + j) * 32 + bk] = bv[j];
    }
    copy_wait();
    __syncthreads();
  }

  // Epilogue. C layout: slot i -> M = i + 8*hi (lanes 0-15 / 16-31), N = lr.
#pragma unroll
  for (int t = 0; t < 4; ++t) {
#pragma unroll
    for (int i = 0; i < 8; ++i) {
      int m = m0 + wave * 16 + i + 8 * hi;
      int n = n0 + t * 16 + lr;
      float v = acc[t][i];
      if (mode == 1) {
        Cout[(size_t)m * HID + n] = v;
      } else {
        int b = m >> 11, s = m & (SEQ - 1);
        if (n < NH * HD) {
          int h = n >> 6, d = n & 63;
          Qb[(((size_t)(b * NH + h)) * SEQ + s) * HD + d] = (bf16_t)v;
        } else if (n < (NH + NKV) * HD) {
          int g = (n - NH * HD) >> 6, d = n & 63;
          Kb[(((size_t)(b * NKV + g)) * SEQ + s) * HD + d] = (bf16_t)v;
        } else {
          int g = (n - (NH + NKV) * HD) >> 6, d = n & 63;
          Vb[(((size_t)(b * NKV + g)) * SEQ + s) * HD + d] = (bf16_t)v;
        }
      }
    }
  }
}

// ---------------------------------------------------------------------------
// Flash attention (causal, GQA rep=4). Block = (b, h, 128 q rows), 8 waves,
// each wave owns 16 q rows. Double-buffered 64-key K/V tiles in LDS: the
// async K copy and V register loads for tile i+1 overlap with tile i's
// QK^T WMMA + softmax + PV WMMA.
// ---------------------------------------------------------------------------
__global__ __launch_bounds__(256)
void attn_fwd_wmma(const bf16_t* __restrict__ Qb, const bf16_t* __restrict__ Kb,
                   const bf16_t* __restrict__ Vb, bf16_t* __restrict__ AOb) {
  __shared__ __align__(16) bf16_t Ksh[2][64 * 64];     // [key][d]   2 x 8 KB
  __shared__ __align__(16) bf16_t Vsh[2][64 * 64];     // [d][key]   2 x 8 KB
  __shared__ __align__(16) bf16_t Psh[8 * 16 * 64];    // per-wave P 16 KB

  const int tid  = threadIdx.x;
  const int wave = tid >> 5;
  const int lane = tid & 31;
  const int lr   = lane & 15;
  const int hi   = lane >> 4;

  const int bh = blockIdx.x;           // 0..B*NH-1
  const int bi = bh >> 4, h = bh & 15;
  const int g  = h >> 2;               // kv group (repeat_interleave: h // 4)
  const int q0 = blockIdx.y * 128;

  const bf16_t* Qbase = Qb + (size_t)(bi * NH + h) * SEQ * HD;
  const bf16_t* Kbase = Kb + (size_t)(bi * NKV + g) * SEQ * HD;
  const bf16_t* Vbase = Vb + (size_t)(bi * NKV + g) * SEQ * HD;

  const int vkey = tid >> 2, vdb = (tid & 3) * 16;     // V-tile staging role

  const int qrow_base = q0 + wave * 16;
  // Q fragments straight from global (A layout is K-contiguous per lane)
  bf16x16 qa0 = ld_frag_a(Qbase + (size_t)(qrow_base + lr) * HD + 0, hi);
  bf16x16 qa1 = ld_frag_a(Qbase + (size_t)(qrow_base + lr) * HD + 32, hi);

  // Prologue: stage kv tile 0 into buffer 0.
  {
    const bf16_t* s = Kbase + tid * 16;
    copy16_g2l(s, &Ksh[0][tid * 16]);
    copy16_g2l(s + 8, &Ksh[0][tid * 16 + 8]);
    const bf16_t* vs = Vbase + (size_t)vkey * HD + vdb;
    bf16x8 v0 = *(const bf16x8*)vs;
    bf16x8 v1 = *(const bf16x8*)(vs + 8);
#pragma unroll
    for (int j = 0; j < 8; ++j) {
      Vsh[0][(vdb + j) * 64 + vkey]     = v0[j];
      Vsh[0][(vdb + 8 + j) * 64 + vkey] = v1[j];
    }
  }
  copy_wait();
  __syncthreads();

  f32x8 oacc[4] = {};
  float run_m[8], run_l[8];
#pragma unroll
  for (int i = 0; i < 8; ++i) { run_m[i] = -1e30f; run_l[i] = 0.0f; }

  const int ntile = 2 * blockIdx.y + 2;  // causal: keys up to q0+127
  for (int it = 0; it < ntile; ++it) {
    const int kv0 = it * 64;
    const int cur = it & 1, nxt = cur ^ 1;
    const bool has_next = (it + 1 < ntile);

    bf16x8 nv0 = {}, nv1 = {};
    if (has_next) {  // issue next-tile staging before computing on `cur`
      const int kvn = kv0 + 64;
      const bf16_t* s = Kbase + (size_t)kvn * HD + tid * 16;
      copy16_g2l(s, &Ksh[nxt][tid * 16]);
      copy16_g2l(s + 8, &Ksh[nxt][tid * 16 + 8]);
      const bf16_t* vs = Vbase + (size_t)(kvn + vkey) * HD + vdb;
      nv0 = *(const bf16x8*)vs;
      nv1 = *(const bf16x8*)(vs + 8);
    }

    // S = Q * K^T  (K-dim = head dim, 2 steps of 32)
    f32x8 sacc[4] = {};
#pragma unroll
    for (int t = 0; t < 4; ++t) {
      bf16x16 kb0 = ld_frag_b(&Ksh[cur][(t * 16 + lr) * 64 + 0], hi);
      sacc[t] = WMMA_BF16(qa0, kb0, sacc[t]);
      bf16x16 kb1 = ld_frag_b(&Ksh[cur][(t * 16 + lr) * 64 + 32], hi);
      sacc[t] = WMMA_BF16(qa1, kb1, sacc[t]);
    }

    // Online softmax. Slot i -> q row (i + 8*hi); col -> key kv0 + t*16 + lr.
#pragma unroll
    for (int i = 0; i < 8; ++i) {
      const int qrow = qrow_base + i + 8 * hi;
      float sv[4];
      float mloc = -1e30f;
#pragma unroll
      for (int t = 0; t < 4; ++t) {
        float s = sacc[t][i] * ATT_SCALE;
        if (kv0 + t * 16 + lr > qrow) s = -1e30f;  // causal mask
        sv[t] = s;
        mloc = fmaxf(mloc, s);
      }
#pragma unroll
      for (int off = 8; off >= 1; off >>= 1) mloc = fmaxf(mloc, __shfl_xor(mloc, off, 32));
      float mnew = fmaxf(run_m[i], mloc);
      float corr = __expf(run_m[i] - mnew);
      run_m[i] = mnew;
      float rs = 0.0f;
#pragma unroll
      for (int t = 0; t < 4; ++t) {
        float p = __expf(sv[t] - mnew);
        Psh[wave * 1024 + (i + 8 * hi) * 64 + t * 16 + lr] = (bf16_t)p;
        rs += p;
      }
#pragma unroll
      for (int off = 8; off >= 1; off >>= 1) rs += __shfl_xor(rs, off, 32);
      run_l[i] = run_l[i] * corr + rs;
#pragma unroll
      for (int t = 0; t < 4; ++t) oacc[t][i] *= corr;
    }

    // O += P * V  (K-dim = keys, 2 steps of 32). DS ops are in-order per wave,
    // so the per-wave Psh write->read needs no workgroup barrier.
#pragma unroll
    for (int kk = 0; kk < 64; kk += 32) {
      bf16x16 pa = ld_frag_a(&Psh[wave * 1024 + lr * 64 + kk], hi);
#pragma unroll
      for (int t = 0; t < 4; ++t) {
        bf16x16 vb = ld_frag_b(&Vsh[cur][(t * 16 + lr) * 64 + kk], hi);
        oacc[t] = WMMA_BF16(pa, vb, oacc[t]);
      }
    }

    if (has_next) {  // transpose-store next V tile into the other buffer
#pragma unroll
      for (int j = 0; j < 8; ++j) {
        Vsh[nxt][(vdb + j) * 64 + vkey]     = nv0[j];
        Vsh[nxt][(vdb + 8 + j) * 64 + vkey] = nv1[j];
      }
    }
    copy_wait();
    __syncthreads();
  }

  // Normalize and store attention output as bf16 [b, s, h*64 + d]
#pragma unroll
  for (int i = 0; i < 8; ++i) {
    float inv = 1.0f / run_l[i];
    int s = qrow_base + i + 8 * hi;
#pragma unroll
    for (int t = 0; t < 4; ++t) {
      int d = t * 16 + lr;
      AOb[((size_t)(bi * SEQ + s)) * HID + h * HD + d] = (bf16_t)(oacc[t][i] * inv);
    }
  }
}

// ---------------------------------------------------------------------------
// Launch
// ---------------------------------------------------------------------------
extern "C" void kernel_launch(void* const* d_in, const int* in_sizes, int n_in,
                              void* d_out, int out_size, void* d_ws, size_t ws_size,
                              hipStream_t stream) {
  const float* hs   = (const float*)d_in[0];  // [B,S,HID]
  const float* Wqkv = (const float*)d_in[1];  // [HID, 1536]
  const float* Wo   = (const float*)d_in[2];  // [HID, HID]
  float* out = (float*)d_out;                 // [B,S,HID]

  // Workspace carve-up (bf16 staging, ~33 MB total, 16B-aligned slices)
  char* w = (char*)d_ws;
  bf16_t* hb    = (bf16_t*)w; w += (size_t)MTOT * HID * sizeof(bf16_t);
  bf16_t* wqkvb = (bf16_t*)w; w += (size_t)HID * NQKV * sizeof(bf16_t);
  bf16_t* wob   = (bf16_t*)w; w += (size_t)HID * HID * sizeof(bf16_t);
  bf16_t* Qb    = (bf16_t*)w; w += (size_t)BATCH * NH * SEQ * HD * sizeof(bf16_t);
  bf16_t* Kb    = (bf16_t*)w; w += (size_t)BATCH * NKV * SEQ * HD * sizeof(bf16_t);
  bf16_t* Vb    = (bf16_t*)w; w += (size_t)BATCH * NKV * SEQ * HD * sizeof(bf16_t);
  bf16_t* AOb   = (bf16_t*)w; w += (size_t)MTOT * HID * sizeof(bf16_t);
  (void)ws_size; (void)in_sizes; (void)n_in; (void)out_size;

  {  // converts
    int n1 = MTOT * HID, n2 = HID * NQKV, n3 = HID * HID;
    f32_to_bf16_kernel<<<(n1 + 255) / 256, 256, 0, stream>>>(hs, hb, n1);
    f32_to_bf16_kernel<<<(n2 + 255) / 256, 256, 0, stream>>>(Wqkv, wqkvb, n2);
    f32_to_bf16_kernel<<<(n3 + 255) / 256, 256, 0, stream>>>(Wo, wob, n3);
  }

  {  // QKV projection with scatter to [b, head, s, d]
    dim3 grid(MTOT / 128, NQKV / 64);
    gemm_bf16_wmma<<<grid, 256, 0, stream>>>(hb, wqkvb, NQKV, 0, Qb, Kb, Vb, nullptr);
  }

  {  // flash attention
    dim3 grid(BATCH * NH, SEQ / 128);
    attn_fwd_wmma<<<grid, 256, 0, stream>>>(Qb, Kb, Vb, AOb);
  }

  {  // output projection -> fp32 d_out
    dim3 grid(MTOT / 128, HID / 64);
    gemm_bf16_wmma<<<grid, 256, 0, stream>>>(AOb, wob, HID, 1, nullptr, nullptr, nullptr, out);
  }
}
